// LGeMSelfAttention_61873298866739
// MI455X (gfx1250) — matrix operand order repaired and checked by reading, not compiled, not run on hardware
//
#include <hip/hip_runtime.h>
#include <math.h>

// ---------------------------------------------------------------------------
// MI455X (gfx1250) self-attention: fused WMMA GEMMs + flash attention.
// All matrix math goes through v_wmma_f32_16x16x32_f16 (wave32).
// Workspace layout (needs 4 * M*C * 2 bytes = 128 MB):
//   Qh | Kh | Vh | AttnH   (all _Float16, [B*T, C] row-major)
// ---------------------------------------------------------------------------

typedef _Float16 v16h __attribute__((ext_vector_type(16)));
typedef _Float16 v8h  __attribute__((ext_vector_type(8)));
typedef _Float16 v4h  __attribute__((ext_vector_type(4)));
typedef float    v8f  __attribute__((ext_vector_type(8)));

#define LDK  40   // flash LDS row stride (32 data + 8 pad halves, 16B aligned)
#define LDKG 72   // gemm LDS row stride (64 data + 8 pad halves, 16B aligned)
#define LDKK 136  // flash K-tile row stride (128 data + 8 pad halves)

static __device__ __forceinline__ v16h pack16(v8h lo, v8h hi) {
    v16h r;
#pragma unroll
    for (int i = 0; i < 8; ++i) { r[i] = lo[i]; r[i + 8] = hi[i]; }
    return r;
}

// A-matrix fragment (16xK=32 f16): lane = M row (lane&15);
// lanes 0-15 hold K={0..7,16..23}, lanes 16-31 hold K={8..15,24..31}.
static __device__ __forceinline__ v16h fragA(const _Float16* base, int stride, int lane) {
    int r = lane & 15;
    int off = (lane & 16) ? 8 : 0;
    const _Float16* p = base + (size_t)r * stride + off;
    return pack16(*(const v8h*)p, *(const v8h*)(p + 16));
}

// B-matrix fragment (K=32x16 f16): lane = N col (lane&15);
// lanes 0-15 hold K={0..7,16..23}, lanes 16-31 hold K={8..15,24..31}.
static __device__ __forceinline__ v16h fragB(const _Float16* base, int stride, int lane) {
    int r = lane & 15;
    int off = (lane & 16) ? 8 : 0;
    const _Float16* p = base + (size_t)r * stride + off;
    return pack16(*(const v8h*)p, *(const v8h*)(p + 16));
}

static __device__ __forceinline__ v8f wmma_f16(v16h a, v16h b, v8f c) {
    return __builtin_amdgcn_wmma_f32_16x16x32_f16(false, a, false, b, (short)0, c, false, false);
}

// ---------------------------------------------------------------------------
// Tiled GEMM: Out[M,N] = A[M,K] * B[K,N]
//   AHALF: 0 = A is fp32 (converted to f16 on LDS stage), 1 = A is f16
//   B always fp32 (weights), converted on LDS stage.
//   MODE: 0 = store f16, 1 = rotary(sin,cos) then store f16, 2 = store fp32
// Block: 256 threads (8 waves). Tile 128x128, BK=64, double-buffered LDS.
// Each wave: 32x64 output = 2x4 WMMA tiles; 16 WMMAs per barrier.
// Pipeline: issue next tile's global loads -> 16 WMMAs on current tile ->
// convert + LDS-store next tile into the other buffer -> one barrier.
// Fragments of each K=32 sub-step preloaded into distinct registers so the
// ds_load clause overlaps the WMMA chain (no per-WMMA dscnt-0 stalls).
// ---------------------------------------------------------------------------
template<int AHALF, int MODE>
__global__ __launch_bounds__(256)
void gemm_wmma(const void* __restrict__ Asrc,
               const float* __restrict__ Bsrc, void* __restrict__ Out,
               int M, int N, int Kd,
               const float* __restrict__ sinp, const float* __restrict__ cosp,
               int Tlen, int Dh) {
    __shared__ _Float16 Ash[2][128 * LDKG];
    __shared__ _Float16 Bsh[2][128 * LDKG];   // stored transposed: Bsh[n][k]

    const int tid  = threadIdx.x;
    const int lane = tid & 31;
    const int wave = tid >> 5;
    const int wy   = wave & 3;      // 4 waves along M
    const int wx   = wave >> 2;     // 2 waves along N
    const int m0   = blockIdx.y * 128;
    const int n0   = blockIdx.x * 128;

    const _Float16* Ag = (const _Float16*)Asrc;
    const float*    Af = (const float*)Asrc;

    // per-thread staging indices (8 chunks of 4 elements each for A and B)
    int aRow[8], aKc[8], bKk[8], bNc[8];
#pragma unroll
    for (int q = 0; q < 8; ++q) {
        int cid = tid + 256 * q;            // 0..2047
        aRow[q] = cid >> 4;                 // 0..127
        aKc[q]  = (cid & 15) * 4;           // 0..60
        bKk[q]  = cid >> 5;                 // 0..63
        bNc[q]  = (cid & 31) * 4;           // 0..124
    }

    v8f acc[2][4];
#pragma unroll
    for (int mt = 0; mt < 2; ++mt)
#pragma unroll
        for (int nt = 0; nt < 4; ++nt)
#pragma unroll
            for (int e = 0; e < 8; ++e) acc[mt][nt][e] = 0.0f;

    float4 aR[8], bR[8];
    v4h    aRh[8];

    // ---- issue global loads for tile at k0 into registers ----
    auto issue_loads = [&](int k0) {
#pragma unroll
        for (int q = 0; q < 8; ++q) {
            if (AHALF)
                aRh[q] = *(const v4h*)(Ag + (size_t)(m0 + aRow[q]) * Kd + k0 + aKc[q]);
            else
                aR[q] = *(const float4*)(Af + (size_t)(m0 + aRow[q]) * Kd + k0 + aKc[q]);
        }
#pragma unroll
        for (int q = 0; q < 8; ++q)
            bR[q] = *(const float4*)(Bsrc + (size_t)(k0 + bKk[q]) * N + n0 + bNc[q]);
    };

    // ---- convert + store staged registers into LDS buffer `buf` ----
    auto store_lds = [&](int buf) {
#pragma unroll
        for (int q = 0; q < 8; ++q) {
            v4h hv;
            if (AHALF) hv = aRh[q];
            else       hv = v4h{ (_Float16)aR[q].x, (_Float16)aR[q].y,
                                 (_Float16)aR[q].z, (_Float16)aR[q].w };
            *((v4h*)&Ash[buf][aRow[q] * LDKG + aKc[q]]) = hv;
        }
#pragma unroll
        for (int q = 0; q < 8; ++q) {
            Bsh[buf][(bNc[q] + 0) * LDKG + bKk[q]] = (_Float16)bR[q].x;
            Bsh[buf][(bNc[q] + 1) * LDKG + bKk[q]] = (_Float16)bR[q].y;
            Bsh[buf][(bNc[q] + 2) * LDKG + bKk[q]] = (_Float16)bR[q].z;
            Bsh[buf][(bNc[q] + 3) * LDKG + bKk[q]] = (_Float16)bR[q].w;
        }
    };

    const int nk = Kd / 64;
    issue_loads(0);
    store_lds(0);

    for (int kt = 0; kt < nk; ++kt) {
        __syncthreads();
        const int buf = kt & 1;
        const bool more = (kt + 1) < nk;
        if (more) issue_loads((kt + 1) * 64);   // global loads in flight over WMMAs

#pragma unroll
        for (int ks = 0; ks < 2; ++ks) {        // two K=32 sub-steps per tile
            // preload all 6 fragments of this sub-step (distinct registers ->
            // ds_load clause overlaps the WMMA chain)
            v16h aF0 = fragA(&Ash[buf][(wy * 32 +  0) * LDKG + ks * 32], LDKG, lane);
            v16h aF1 = fragA(&Ash[buf][(wy * 32 + 16) * LDKG + ks * 32], LDKG, lane);
            v16h bFr[4];
#pragma unroll
            for (int nt = 0; nt < 4; ++nt)
                bFr[nt] = fragB(&Bsh[buf][(wx * 64 + nt * 16) * LDKG + ks * 32], LDKG, lane);
#pragma unroll
            for (int nt = 0; nt < 4; ++nt) {
                acc[0][nt] = wmma_f16(aF0, bFr[nt], acc[0][nt]);
                acc[1][nt] = wmma_f16(aF1, bFr[nt], acc[1][nt]);
            }
        }

        if (more) store_lds(buf ^ 1);           // other buffer: no barrier needed
    }

    // ---- epilogue ----
    const int colH = lane & 15;
    const int hiH  = (lane >> 4) & 1;   // C/D: VGPR i => row i + 8*hiH
#pragma unroll
    for (int mt = 0; mt < 2; ++mt) {
#pragma unroll
        for (int nt = 0; nt < 4; ++nt) {
            int rb = m0 + wy * 32 + mt * 16 + hiH * 8;
            int cb = n0 + wx * 64 + nt * 16 + colH;
            v8f a = acc[mt][nt];
#pragma unroll
            for (int i = 0; i < 8; ++i) {
                int r = rb + i;
                float v = a[i];
                if (MODE == 1) {
                    // GPT-J interleaved rotary; pair column lives in lane^1
                    float partner = __shfl_xor(v, 1, 32);
                    int t = r % Tlen;
                    int d = cb & (Dh - 1);
                    float cc = cosp[t * Dh + d];
                    float ss = sinp[t * Dh + d];
                    float o = (cb & 1) ? (v * cc + partner * ss)
                                       : (v * cc - partner * ss);
                    ((_Float16*)Out)[(size_t)r * N + cb] = (_Float16)o;
                } else if (MODE == 0) {
                    ((_Float16*)Out)[(size_t)r * N + cb] = (_Float16)v;
                } else {
                    ((float*)Out)[(size_t)r * N + cb] = v;
                }
            }
        }
    }
}

// ---------------------------------------------------------------------------
// Flash attention (causal). Block = 128 threads (4 waves), one (b,h) head and
// a 64-row query tile per block; each wave owns 16 query rows. Keys processed
// 32 at a time (matches WMMA K=32). D = 128.
// K and V tiles staged once per block in LDS.  The softmax row-sum is
// computed by the matrix pipe (P x ones WMMA) instead of lane shuffles.
// ---------------------------------------------------------------------------
__global__ __launch_bounds__(128)
void flash_attn(const _Float16* __restrict__ Qh, const _Float16* __restrict__ Kh,
                const _Float16* __restrict__ Vh, _Float16* __restrict__ AttnH,
                int Hh, int Tt, int Cc) {
    const int D = 128;
    __shared__ _Float16 Kt[32 * LDKK];         // K tile key-major: Kt[key][d]
    __shared__ _Float16 Vt[128 * LDK];         // V tile transposed: Vt[d][key]
    __shared__ _Float16 Plds[4 * 16 * LDK];    // per-wave P staging 16x32

    const int bh = blockIdx.x;
    const int b  = bh / Hh;
    const int h  = bh % Hh;
    const int q0 = blockIdx.y * 64;
    const int tid  = threadIdx.x;
    const int wave = tid >> 5;
    const int lane = tid & 31;
    const int qw   = q0 + wave * 16;
    const int colH = lane & 15;
    const int hiH  = (lane >> 4) & 1;
    const size_t headBase = (size_t)(b * Tt) * Cc + (size_t)h * D;

    // Q fragments (A-matrix): 4 chunks of K-dim (d = 0..127)
    v16h qf[4];
#pragma unroll
    for (int c = 0; c < 4; ++c)
        qf[c] = fragA(Qh + headBase + (size_t)qw * Cc + c * 32, Cc, lane);

    // all-ones B fragment: P(16x32) x ones(32x16) -> row sums in every column
    v16h onesF;
#pragma unroll
    for (int i = 0; i < 16; ++i) onesF[i] = (_Float16)1.0f;

    float mI[8], lI[8];
    v8f O[8];
#pragma unroll
    for (int i = 0; i < 8; ++i) { mI[i] = -1e30f; lI[i] = 0.0f; }
#pragma unroll
    for (int c = 0; c < 8; ++c)
#pragma unroll
        for (int e = 0; e < 8; ++e) O[c][e] = 0.0f;

    const float scale = 0.0883883476483184f;   // 1/sqrt(128)
    const int ntiles = q0 / 32 + 2;            // causal: keys <= q0+63

    const int sKey  = tid & 31;
    const int sDgrp = tid >> 5;                // 4 groups of 32 d-values

    for (int kt = 0; kt < ntiles; ++kt) {
        const int kb = kt * 32;
        // prefetch next tile's K/V rows while we work on this one
        if (kt + 1 < ntiles) {
            size_t nb = headBase + (size_t)(kb + 32 + sKey) * Cc + sDgrp * 32;
            __builtin_prefetch((const char*)(Vh + nb), 0, 1);
            __builtin_prefetch((const char*)(Kh + nb), 0, 1);
        }
        __syncthreads();
        // ---- stage K (key-major) and V (transposed) tiles into LDS ----
        {
            size_t grow = headBase + (size_t)(kb + sKey) * Cc + sDgrp * 32;
            v8h kreg[4], vreg[4];
#pragma unroll
            for (int g = 0; g < 4; ++g) {      // all 8 loads in flight first
                kreg[g] = *(const v8h*)(Kh + grow + g * 8);
                vreg[g] = *(const v8h*)(Vh + grow + g * 8);
            }
#pragma unroll
            for (int g = 0; g < 4; ++g)        // K: contiguous b128 LDS stores
                *(v8h*)&Kt[sKey * LDKK + sDgrp * 32 + g * 8] = kreg[g];
#pragma unroll
            for (int g = 0; g < 4; ++g)        // V: transpose scatter
#pragma unroll
                for (int j = 0; j < 8; ++j)
                    Vt[(sDgrp * 32 + g * 8 + j) * LDK + sKey] = vreg[g][j];
        }
        __syncthreads();

        // ---- scores S = Q K^T for 32 keys (two 16-col WMMA tiles) ----
        v8f S[2];
#pragma unroll
        for (int j = 0; j < 2; ++j) {
            // preload all 4 K fragments, then run the WMMA chain
            v16h kfr[4];
#pragma unroll
            for (int c = 0; c < 4; ++c)
                kfr[c] = fragB(&Kt[(j * 16) * LDKK + c * 32], LDKK, lane);
            v8f s = {0.f, 0.f, 0.f, 0.f, 0.f, 0.f, 0.f, 0.f};
#pragma unroll
            for (int c = 0; c < 4; ++c)
                s = wmma_f16(qf[c], kfr[c], s);

            int key = kb + j * 16 + colH;
#pragma unroll
            for (int i = 0; i < 8; ++i) {
                int row = qw + i + hiH * 8;
                float sv = s[i] * scale;
                if (key > row) sv = -1e9f;     // causal mask (matches ref bias)
                s[i] = sv;
            }
            S[j] = s;
        }

        // ---- online softmax update ----
        float rm[8];
#pragma unroll
        for (int i = 0; i < 8; ++i) rm[i] = fmaxf(S[0][i], S[1][i]);
#pragma unroll
        for (int ms = 1; ms < 16; ms <<= 1)
#pragma unroll
            for (int i = 0; i < 8; ++i) rm[i] = fmaxf(rm[i], __shfl_xor(rm[i], ms, 32));

        float resc[8];
#pragma unroll
        for (int i = 0; i < 8; ++i) {
            float mn = fmaxf(mI[i], rm[i]);
            resc[i] = __expf(mI[i] - mn);
            mI[i] = mn;
        }
#pragma unroll
        for (int j = 0; j < 2; ++j)
#pragma unroll
            for (int i = 0; i < 8; ++i) S[j][i] = __expf(S[j][i] - mI[i]);

#pragma unroll
        for (int c = 0; c < 8; ++c)
#pragma unroll
            for (int i = 0; i < 8; ++i) O[c][i] *= resc[i];

        // ---- P (D-layout) -> LDS -> A-layout fragment ----
        _Float16* pw = &Plds[wave * 16 * LDK];
#pragma unroll
        for (int j = 0; j < 2; ++j)
#pragma unroll
            for (int i = 0; i < 8; ++i)
                pw[(i + hiH * 8) * LDK + j * 16 + colH] = (_Float16)S[j][i];

        v16h pf = fragA(pw, LDK, lane);

        // ---- row sums via the matrix pipe: P x ones -> each col = row sum ----
        {
            v8f z = {0.f, 0.f, 0.f, 0.f, 0.f, 0.f, 0.f, 0.f};
            v8f srow = wmma_f16(pf, onesF, z);
#pragma unroll
            for (int i = 0; i < 8; ++i) lI[i] = lI[i] * resc[i] + srow[i];
        }

        // ---- O += P @ V (8 d-chunks of 16), fragments preloaded in halves ----
#pragma unroll
        for (int half = 0; half < 2; ++half) {
            v16h vfr[4];
#pragma unroll
            for (int c4 = 0; c4 < 4; ++c4)
                vfr[c4] = fragB(&Vt[((half * 4 + c4) * 16) * LDK], LDK, lane);
#pragma unroll
            for (int c4 = 0; c4 < 4; ++c4) {
                int c = half * 4 + c4;
                O[c] = wmma_f16(pf, vfr[c4], O[c]);
            }
        }
    }

    // ---- normalize and store (f16, [B*T, C]) ----
#pragma unroll
    for (int i = 0; i < 8; ++i) lI[i] = 1.0f / lI[i];
#pragma unroll
    for (int c = 0; c < 8; ++c)
#pragma unroll
        for (int i = 0; i < 8; ++i) {
            int row = qw + i + hiH * 8;
            AttnH[headBase + (size_t)row * Cc + c * 16 + colH] = (_Float16)(O[c][i] * lI[i]);
        }
}

// ---------------------------------------------------------------------------
extern "C" void kernel_launch(void* const* d_in, const int* in_sizes, int n_in,
                              void* d_out, int out_size, void* d_ws, size_t ws_size,
                              hipStream_t stream) {
    (void)in_sizes; (void)n_in; (void)out_size; (void)ws_size;
    const int Bb = 2, Tt = 2048, Cc = 4096, Hh = 32, Dd = 128;
    const int M = Bb * Tt;

    const float* X    = (const float*)d_in[0];
    const float* sinp = (const float*)d_in[1];
    const float* cosp = (const float*)d_in[2];
    /* d_in[3] mask_bias: causal mask applied analytically in-kernel */
    const float* Wq   = (const float*)d_in[4];
    const float* Wk   = (const float*)d_in[5];
    const float* Wv   = (const float*)d_in[6];
    const float* Wo   = (const float*)d_in[7];
    float* out = (float*)d_out;

    const size_t per = (size_t)M * Cc;          // 16.7M elements
    _Float16* Qh = (_Float16*)d_ws;             // needs 4*per*2 = 128 MB ws
    _Float16* Kh = Qh + per;
    _Float16* Vh = Kh + per;
    _Float16* Ah = Vh + per;

    dim3 gg(Cc / 128, M / 128);
    dim3 bb(256);
    // Q/K projections with fused rotary, V projection plain (f16 outputs)
    gemm_wmma<0, 1><<<gg, bb, 0, stream>>>(X, Wq, Qh, M, Cc, Cc, sinp, cosp, Tt, Dd);
    gemm_wmma<0, 1><<<gg, bb, 0, stream>>>(X, Wk, Kh, M, Cc, Cc, sinp, cosp, Tt, Dd);
    gemm_wmma<0, 0><<<gg, bb, 0, stream>>>(X, Wv, Vh, M, Cc, Cc, sinp, cosp, Tt, Dd);
    // causal flash attention
    flash_attn<<<dim3(Bb * Hh, Tt / 64), dim3(128), 0, stream>>>(Qh, Kh, Vh, Ah, Hh, Tt, Cc);
    // output projection, fp32 result
    gemm_wmma<1, 2><<<gg, bb, 0, stream>>>(Ah, Wo, out, M, Cc, Cc, sinp, cosp, Tt, Dd);
}
